// MultiHeadLatentAttention_57217554317357
// MI455X (gfx1250) — compile-verified, matrix operand order checked by reading, not run on hardware
//
#include <hip/hip_runtime.h>

// ---------- types ----------
typedef __attribute__((ext_vector_type(16))) __bf16 v16bf;
typedef __attribute__((ext_vector_type(8)))  float  v8f;
typedef unsigned short ushort_t;

struct alignas(16) U4 { unsigned int x, y, z, w; };

union Frag {
    U4        u[2];
    ushort_t  h[16];
    v16bf     v;
};

union B16x16 { U4 u[2]; ushort_t h[16]; };

__device__ __forceinline__ ushort_t f2bf(float f) {
    union { float f; unsigned u; } x; x.f = f;
    unsigned r = x.u + 0x7fffu + ((x.u >> 16) & 1u);
    return (ushort_t)(r >> 16);
}

// ---------- problem constants ----------
#define B_   2
#define S_   2048
#define H_   16
#define DM_  2048
#define QL_  1536
#define KVL_ 512
#define NOPE_ 128
#define ROPE_ 64
#define QKHD_ 192
#define VHD_  128
#define BS_  4096   // B_*S_

// =====================================================================
// cast f32 -> bf16
// =====================================================================
__global__ void mla_cast_f32_bf16(const float* __restrict__ a,
                                  ushort_t* __restrict__ o, long n) {
    long i = (long)blockIdx.x * blockDim.x + threadIdx.x;
    long stride = (long)gridDim.x * blockDim.x;
    for (; i < n; i += stride) o[i] = f2bf(a[i]);
}

// =====================================================================
// GEMM: C(f32, MxN) = A(bf16, MxK row-major) * B(bf16, KxN row-major)
// block tile 128x128, K-step 32, 8 waves, double-buffered LDS.
// Each wave owns a 16x128 strip: 8 WMMAs per K-step, fragments batch-
// loaded (one dscnt wait) so the WMMA chain issues back-to-back.
// =====================================================================
#define BM 128
#define BN 128
#define BK 32
#define LDP 40   // padded LDS row stride (ushorts), 80B = multiple of 16B

__global__ __launch_bounds__(256)
void mla_gemm_bf16(const ushort_t* __restrict__ A,
                   const ushort_t* __restrict__ Bm,
                   float* __restrict__ C,
                   int M, int N, int Kd) {
    __shared__ ushort_t As[2][BM][LDP];
    __shared__ ushort_t Bs[2][BN][LDP];

    const int tid  = threadIdx.x;
    const int lane = tid & 31;
    const int wave = tid >> 5;
    const int bm = blockIdx.y * BM;
    const int bn = blockIdx.x * BN;

    const int mi  = wave;             // wave's 16-row M sub-tile
    const int r   = lane & 15;
    const int hi  = lane >> 4;
    const int off = hi * 8;

    v8f acc[8];
    for (int j = 0; j < 8; ++j)
        for (int e = 0; e < 8; ++e) acc[j][e] = 0.0f;

    // staging assignment
    const int aRow = tid >> 1;           // 0..127
    const int aCol = (tid & 1) * 16;     // 0 or 16
    const int bRow = tid >> 3;           // 0..31
    const int bCol = (tid & 7) * 16;     // 0..112
    const bool edgeN = (bn + BN > N);

    // ---- prologue: stage tile 0 into buffer 0 ----
    {
        const ushort_t* ap = A + (size_t)(bm + aRow) * Kd + aCol;
        U4 av0, av1;
        av0.x = av0.y = av0.z = av0.w = 0u; av1 = av0;
        if (bm + aRow < M) { av0 = *(const U4*)ap; av1 = *(const U4*)(ap + 8); }
        *(U4*)&As[0][aRow][aCol]     = av0;
        *(U4*)&As[0][aRow][aCol + 8] = av1;

        const ushort_t* bp = Bm + (size_t)bRow * N + bn + bCol;
        B16x16 bt;
        if (!edgeN) {
            bt.u[0] = *(const U4*)bp; bt.u[1] = *(const U4*)(bp + 8);
        } else {
            #pragma unroll
            for (int i = 0; i < 16; ++i) {
                int n = bn + bCol + i;
                bt.h[i] = (n < N) ? bp[i] : (ushort_t)0;
            }
        }
        #pragma unroll
        for (int i = 0; i < 16; ++i) Bs[0][bCol + i][bRow] = bt.h[i];
    }
    __syncthreads();

    const int nk = Kd / BK;
    for (int ki = 0; ki < nk; ++ki) {
        const int c = ki & 1;
        const bool hasNext = (ki + 1 < nk);

        // ---- issue next tile's global loads (latency hides behind WMMA) ----
        U4 av0, av1; B16x16 bt;
        if (hasNext) {
            const int k0n = (ki + 1) * BK;
            const ushort_t* ap = A + (size_t)(bm + aRow) * Kd + k0n + aCol;
            av0.x = av0.y = av0.z = av0.w = 0u; av1 = av0;
            if (bm + aRow < M) { av0 = *(const U4*)ap; av1 = *(const U4*)(ap + 8); }
            const ushort_t* bp = Bm + (size_t)(k0n + bRow) * N + bn + bCol;
            if (!edgeN) {
                bt.u[0] = *(const U4*)bp; bt.u[1] = *(const U4*)(bp + 8);
                __builtin_prefetch(ap + BK, 0, 1);
                __builtin_prefetch(bp + (size_t)BK * N, 0, 1);
            } else {
                #pragma unroll
                for (int i = 0; i < 16; ++i) {
                    int n = bn + bCol + i;
                    bt.h[i] = (n < N) ? bp[i] : (ushort_t)0;
                }
            }
        }

        // ---- compute from current buffer: batch fragment loads ----
        Frag a;
        a.u[0] = *(const U4*)&As[c][mi * 16 + r][off];
        a.u[1] = *(const U4*)&As[c][mi * 16 + r][off + 16];
        Frag bfr[8];
        #pragma unroll
        for (int j = 0; j < 8; ++j) {
            bfr[j].u[0] = *(const U4*)&Bs[c][j * 16 + r][off];
            bfr[j].u[1] = *(const U4*)&Bs[c][j * 16 + r][off + 16];
        }
        #pragma unroll
        for (int j = 0; j < 8; ++j)
            acc[j] = __builtin_amdgcn_wmma_f32_16x16x32_bf16(
                false, a.v, false, bfr[j].v, (short)0, acc[j], false, false);

        // ---- write next tile into the other buffer ----
        if (hasNext) {
            const int n2 = c ^ 1;
            *(U4*)&As[n2][aRow][aCol]     = av0;
            *(U4*)&As[n2][aRow][aCol + 8] = av1;
            #pragma unroll
            for (int i = 0; i < 16; ++i) Bs[n2][bCol + i][bRow] = bt.h[i];
        }
        __syncthreads();
    }

    // ---- store C (f32) ----
    #pragma unroll
    for (int j = 0; j < 8; ++j) {
        int col = bn + j * 16 + r;
        if (col < N) {
            #pragma unroll
            for (int e = 0; e < 8; ++e) {
                int row = bm + mi * 16 + e + hi * 8;
                if (row < M) C[(size_t)row * N + col] = acc[j][e];
            }
        }
    }
}

// =====================================================================
// row LayerNorm (f32 in -> bf16 out)
// =====================================================================
__global__ __launch_bounds__(256)
void mla_layernorm(const float* __restrict__ in, int inStride, int L,
                   const float* __restrict__ w, const float* __restrict__ bias,
                   ushort_t* __restrict__ out, int outStride) {
    const int row = blockIdx.x;
    const float* x = in + (size_t)row * inStride;
    ushort_t* o = out + (size_t)row * outStride;

    float s = 0.0f, s2 = 0.0f;
    for (int i = threadIdx.x; i < L; i += blockDim.x) {
        float v = x[i]; s += v; s2 += v * v;
    }
    for (int d = 1; d < 32; d <<= 1) {
        s  += __shfl_xor(s, d, 32);
        s2 += __shfl_xor(s2, d, 32);
    }
    __shared__ float sh[2][8];
    int wave = threadIdx.x >> 5, lane = threadIdx.x & 31;
    if (lane == 0) { sh[0][wave] = s; sh[1][wave] = s2; }
    __syncthreads();
    if (wave == 0) {
        float a  = (lane < 8) ? sh[0][lane] : 0.0f;
        float b2 = (lane < 8) ? sh[1][lane] : 0.0f;
        for (int d = 1; d < 8; d <<= 1) {
            a  += __shfl_xor(a, d, 32);
            b2 += __shfl_xor(b2, d, 32);
        }
        if (lane == 0) { sh[0][0] = a; sh[1][0] = b2; }
    }
    __syncthreads();
    float mean = sh[0][0] / (float)L;
    float var  = sh[1][0] / (float)L - mean * mean;
    float rstd = rsqrtf(var + 1e-5f);
    for (int i = threadIdx.x; i < L; i += blockDim.x)
        o[i] = f2bf((x[i] - mean) * rstd * w[i] + bias[i]);
}

// =====================================================================
// RoPE on Q, scatter to [B,H,S,192] bf16
// =====================================================================
__global__ void mla_rope_q(const float* __restrict__ qraw,
                           const float* __restrict__ cosb,
                           const float* __restrict__ sinb,
                           ushort_t* __restrict__ qb) {
    const long total = (long)BS_ * H_ * QKHD_;
    long idx = (long)blockIdx.x * blockDim.x + threadIdx.x;
    long stride = (long)gridDim.x * blockDim.x;
    for (; idx < total; idx += stride) {
        int d = (int)(idx % QKHD_);
        long t = idx / QKHD_;
        int h = (int)(t % H_); t /= H_;
        int s = (int)(t % S_);
        int b = (int)(t / S_);
        const float* rowp = qraw + ((size_t)(b * S_ + s)) * (H_ * QKHD_) + h * QKHD_;
        float v = rowp[d];
        if (d >= NOPE_) {
            int j = d - NOPE_;
            float c  = cosb[s * ROPE_ + j];
            float sn = sinb[s * ROPE_ + j];
            float other = rowp[NOPE_ + ((j < 32) ? j + 32 : j - 32)];
            float rot = (j < 32) ? -other : other;
            v = v * c + rot * sn;
        }
        qb[(((size_t)(b * H_ + h)) * S_ + s) * QKHD_ + d] = f2bf(v);
    }
}

// =====================================================================
// build K (nope || roped rope, broadcast over heads) and V, bf16
// =====================================================================
__global__ void mla_build_kv(const float* __restrict__ kvbraw,
                             const float* __restrict__ kvaraw,
                             const float* __restrict__ cosb,
                             const float* __restrict__ sinb,
                             ushort_t* __restrict__ kb,
                             ushort_t* __restrict__ vb) {
    const long total = (long)BS_ * H_ * (QKHD_ + VHD_); // 320 per (b,s,h)
    long idx = (long)blockIdx.x * blockDim.x + threadIdx.x;
    long stride = (long)gridDim.x * blockDim.x;
    for (; idx < total; idx += stride) {
        int d = (int)(idx % (QKHD_ + VHD_));
        long t = idx / (QKHD_ + VHD_);
        int h = (int)(t % H_); t /= H_;
        int s = (int)(t % S_);
        int b = (int)(t / S_);
        size_t rowBS = (size_t)(b * S_ + s);
        if (d < NOPE_) {
            float v = kvbraw[rowBS * (H_ * (NOPE_ + VHD_)) + h * (NOPE_ + VHD_) + d];
            kb[(((size_t)(b * H_ + h)) * S_ + s) * QKHD_ + d] = f2bf(v);
        } else if (d < QKHD_) {
            int j = d - NOPE_;
            const float* rp = kvaraw + rowBS * (KVL_ + ROPE_) + KVL_;
            float x0 = rp[j];
            float other = rp[(j < 32) ? j + 32 : j - 32];
            float rot = (j < 32) ? -other : other;
            float c  = cosb[s * ROPE_ + j];
            float sn = sinb[s * ROPE_ + j];
            float v = x0 * c + rot * sn;
            kb[(((size_t)(b * H_ + h)) * S_ + s) * QKHD_ + d] = f2bf(v);
        } else {
            int j = d - QKHD_;
            float v = kvbraw[rowBS * (H_ * (NOPE_ + VHD_)) + h * (NOPE_ + VHD_) + NOPE_ + j];
            vb[(((size_t)(b * H_ + h)) * S_ + s) * VHD_ + j] = f2bf(v);
        }
    }
}

// =====================================================================
// causal flash attention: one wave per 16-row Q tile, 32 keys per chunk
// Q,K: [B,H,S,192] bf16 ; V: [B,H,S,128] bf16 ; out: [B,S,H*128] bf16
// scores: two 16x16 tiles (12 WMMAs); PV: 8 full K=32 WMMAs fed by
// LDS-transposed V and a 16x32 P tile (all b128 fragment reads).
// =====================================================================
#define WPB 4
#define VLP 40   // padded Vt row stride

__global__ __launch_bounds__(32 * WPB)
void mla_flash_attn(const ushort_t* __restrict__ Q,
                    const ushort_t* __restrict__ Kp,
                    const ushort_t* __restrict__ Vp,
                    ushort_t* __restrict__ O, int S) {
    const int lane = threadIdx.x & 31;
    const int wave = threadIdx.x >> 5;
    const int bh = blockIdx.y;
    const int b = bh >> 4;
    const int h = bh & 15;
    const int qt = blockIdx.x * WPB + wave;
    const int qrow0 = qt * 16;

    const ushort_t* Qh = Q  + (size_t)bh * S * QKHD_;
    const ushort_t* Kh = Kp + (size_t)bh * S * QKHD_;
    const ushort_t* Vh = Vp + (size_t)bh * S * VHD_;

    const int r  = lane & 15;
    const int hi = lane >> 4;
    const int off = hi * 8;
    const int hi8 = hi * 8;

    __shared__ ushort_t Psh[WPB][16][32];
    __shared__ ushort_t Vt[WPB][VHD_][VLP];   // V transposed: [n][k]

    // preload full Q tile fragments: 6 K-steps of 32 (head dim 192)
    Frag qf[6];
    const ushort_t* qbase = Qh + (size_t)(qrow0 + r) * QKHD_;
    #pragma unroll
    for (int ks = 0; ks < 6; ++ks) {
        qf[ks].u[0] = *(const U4*)(qbase + ks * 32 + off);
        qf[ks].u[1] = *(const U4*)(qbase + ks * 32 + off + 16);
    }

    v8f o8[8];
    for (int j = 0; j < 8; ++j)
        for (int e = 0; e < 8; ++e) o8[j][e] = 0.0f;
    float m_[8], l_[8];
    for (int e = 0; e < 8; ++e) { m_[e] = -3.0e38f; l_[e] = 0.0f; }

    const float scale = 0.0721687836f; // 192^-0.5
    const int nch = (qrow0 + 47) >> 5; // chunks of 32 keys covering diag

    for (int kt = 0; kt < nch; ++kt) {
        const int kc0 = kt * 32;

        // ---- scores: two 16x16 tiles, fragments batch-loaded ----
        v8f s0, s1;
        for (int e = 0; e < 8; ++e) { s0[e] = 0.0f; s1[e] = 0.0f; }
        {
            const ushort_t* kb0 = Kh + (size_t)(kc0 + r) * QKHD_;
            Frag kf[6];
            #pragma unroll
            for (int ks = 0; ks < 6; ++ks) {
                kf[ks].u[0] = *(const U4*)(kb0 + ks * 32 + off);
                kf[ks].u[1] = *(const U4*)(kb0 + ks * 32 + off + 16);
            }
            #pragma unroll
            for (int ks = 0; ks < 6; ++ks)
                s0 = __builtin_amdgcn_wmma_f32_16x16x32_bf16(
                    false, qf[ks].v, false, kf[ks].v, (short)0, s0, false, false);
            const ushort_t* kb1 = Kh + (size_t)(kc0 + 16 + r) * QKHD_;
            #pragma unroll
            for (int ks = 0; ks < 6; ++ks) {
                kf[ks].u[0] = *(const U4*)(kb1 + ks * 32 + off);
                kf[ks].u[1] = *(const U4*)(kb1 + ks * 32 + off + 16);
            }
            #pragma unroll
            for (int ks = 0; ks < 6; ++ks)
                s1 = __builtin_amdgcn_wmma_f32_16x16x32_bf16(
                    false, qf[ks].v, false, kf[ks].v, (short)0, s1, false, false);
        }

        // ---- stage V chunk (32x128) transposed into LDS [n][k] ----
        {
            const ushort_t* vrow = Vh + (size_t)(kc0 + lane) * VHD_;
            #pragma unroll
            for (int g = 0; g < 16; ++g) {
                B16x16 t2; t2.u[0] = *(const U4*)(vrow + g * 8);
                #pragma unroll
                for (int i = 0; i < 8; ++i)
                    Vt[wave][g * 8 + i][lane] = t2.h[i];
            }
        }

        // ---- online softmax over 32 columns ----
        const bool last = (kt == nch - 1);
        #pragma unroll
        for (int e = 0; e < 8; ++e) {
            float a0 = s0[e] * scale;
            float a1 = s1[e] * scale;
            const int qrow = qrow0 + e + hi8;
            if (last) {
                if (kc0 + r > qrow)      a0 = -3.0e38f;
                if (kc0 + 16 + r > qrow) a1 = -3.0e38f;
            }
            float mx = fmaxf(a0, a1);
            for (int d2 = 1; d2 < 16; d2 <<= 1)
                mx = fmaxf(mx, __shfl_xor(mx, d2, 32));
            float mnew = fmaxf(m_[e], mx);
            float pe0 = __expf(a0 - mnew);
            float pe1 = __expf(a1 - mnew);
            float alpha = __expf(m_[e] - mnew);
            m_[e] = mnew;
            float sum = pe0 + pe1;
            for (int d2 = 1; d2 < 16; d2 <<= 1)
                sum += __shfl_xor(sum, d2, 32);
            l_[e] = l_[e] * alpha + sum;
            #pragma unroll
            for (int j = 0; j < 8; ++j) o8[j][e] *= alpha;
            Psh[wave][e + hi8][r]      = f2bf(pe0);
            Psh[wave][e + hi8][16 + r] = f2bf(pe1);
        }
        asm volatile("s_wait_dscnt 0" ::: "memory"); // same-wave LDS RAW

        // ---- P @ V: full K=32 WMMAs, fragments batch-loaded ----
        Frag pf;
        pf.u[0] = *(const U4*)&Psh[wave][r][off];
        pf.u[1] = *(const U4*)&Psh[wave][r][off + 16];
        Frag vfr[8];
        #pragma unroll
        for (int j = 0; j < 8; ++j) {
            vfr[j].u[0] = *(const U4*)&Vt[wave][j * 16 + r][off];
            vfr[j].u[1] = *(const U4*)&Vt[wave][j * 16 + r][off + 16];
        }
        #pragma unroll
        for (int j = 0; j < 8; ++j)
            o8[j] = __builtin_amdgcn_wmma_f32_16x16x32_bf16(
                false, pf.v, false, vfr[j].v, (short)0, o8[j], false, false);
    }

    // ---- epilogue: normalize, write bf16 to [B,S,H*128] ----
    #pragma unroll
    for (int e = 0; e < 8; ++e) {
        int srow = qrow0 + e + hi8;
        float inv = 1.0f / l_[e];
        size_t orow = ((size_t)b * S + srow) * (H_ * VHD_) + (size_t)h * VHD_;
        #pragma unroll
        for (int j = 0; j < 8; ++j)
            O[orow + j * 16 + r] = f2bf(o8[j][e] * inv);
    }
}

// =====================================================================
// host-side launch
// =====================================================================
extern "C" void kernel_launch(void* const* d_in, const int* in_sizes, int n_in,
                              void* d_out, int out_size, void* d_ws, size_t ws_size,
                              hipStream_t stream) {
    (void)in_sizes; (void)n_in; (void)out_size; (void)ws_size;

    const float* x        = (const float*)d_in[0];
    const float* cosb     = (const float*)d_in[1];
    const float* sinb     = (const float*)d_in[2];
    const float* q_a_w    = (const float*)d_in[3];
    const float* q_a_ln_w = (const float*)d_in[4];
    const float* q_a_ln_b = (const float*)d_in[5];
    const float* q_b_w    = (const float*)d_in[6];
    const float* kv_a_w   = (const float*)d_in[7];
    const float* kv_a_ln_w= (const float*)d_in[8];
    const float* kv_a_ln_b= (const float*)d_in[9];
    const float* kv_b_w   = (const float*)d_in[10];
    const float* o_w      = (const float*)d_in[11];
    float* out = (float*)d_out;

    char* p = (char*)d_ws;
    auto carve = [&](size_t bytes) -> char* {
        char* r = p;
        p += (bytes + 255) & ~(size_t)255;
        return r;
    };

    ushort_t* xb      = (ushort_t*)carve((size_t)BS_ * DM_ * 2);
    ushort_t* wqa     = (ushort_t*)carve((size_t)DM_ * QL_ * 2);
    ushort_t* wqb     = (ushort_t*)carve((size_t)QL_ * (H_ * QKHD_) * 2);
    ushort_t* wkva    = (ushort_t*)carve((size_t)DM_ * (KVL_ + ROPE_) * 2);
    ushort_t* wkvb    = (ushort_t*)carve((size_t)KVL_ * (H_ * (NOPE_ + VHD_)) * 2);
    ushort_t* wo      = (ushort_t*)carve((size_t)(H_ * VHD_) * DM_ * 2);
    ushort_t* qlat_b  = (ushort_t*)carve((size_t)BS_ * QL_ * 2);
    ushort_t* ckv_b   = (ushort_t*)carve((size_t)BS_ * KVL_ * 2);
    ushort_t* qb      = (ushort_t*)carve((size_t)BS_ * H_ * QKHD_ * 2);
    ushort_t* kb      = (ushort_t*)carve((size_t)BS_ * H_ * QKHD_ * 2);
    ushort_t* vb      = (ushort_t*)carve((size_t)BS_ * H_ * VHD_ * 2);
    ushort_t* attn_b  = (ushort_t*)carve((size_t)BS_ * (H_ * VHD_) * 2);
    float* qa_raw  = (float*)carve((size_t)BS_ * QL_ * 4);
    float* q_raw   = (float*)carve((size_t)BS_ * H_ * QKHD_ * 4);
    float* kva_raw = (float*)carve((size_t)BS_ * (KVL_ + ROPE_) * 4);
    float* kvb_raw = (float*)carve((size_t)BS_ * H_ * (NOPE_ + VHD_) * 4);

    // ---- casts to bf16 ----
    mla_cast_f32_bf16<<<2048, 256, 0, stream>>>(x, xb, (long)BS_ * DM_);
    mla_cast_f32_bf16<<<1024, 256, 0, stream>>>(q_a_w, wqa, (long)DM_ * QL_);
    mla_cast_f32_bf16<<<1024, 256, 0, stream>>>(q_b_w, wqb, (long)QL_ * H_ * QKHD_);
    mla_cast_f32_bf16<<<512, 256, 0, stream>>>(kv_a_w, wkva, (long)DM_ * (KVL_ + ROPE_));
    mla_cast_f32_bf16<<<512, 256, 0, stream>>>(kv_b_w, wkvb, (long)KVL_ * H_ * (NOPE_ + VHD_));
    mla_cast_f32_bf16<<<1024, 256, 0, stream>>>(o_w, wo, (long)DM_ * DM_);

    // ---- q path ----
    mla_gemm_bf16<<<dim3(QL_ / BN, BS_ / BM), 256, 0, stream>>>(
        xb, wqa, qa_raw, BS_, QL_, DM_);
    mla_layernorm<<<BS_, 256, 0, stream>>>(
        qa_raw, QL_, QL_, q_a_ln_w, q_a_ln_b, qlat_b, QL_);
    mla_gemm_bf16<<<dim3((H_ * QKHD_) / BN, BS_ / BM), 256, 0, stream>>>(
        qlat_b, wqb, q_raw, BS_, H_ * QKHD_, QL_);

    // ---- kv path ----
    mla_gemm_bf16<<<dim3((KVL_ + ROPE_ + BN - 1) / BN, BS_ / BM), 256, 0, stream>>>(
        xb, wkva, kva_raw, BS_, KVL_ + ROPE_, DM_);
    mla_layernorm<<<BS_, 256, 0, stream>>>(
        kva_raw, KVL_ + ROPE_, KVL_, kv_a_ln_w, kv_a_ln_b, ckv_b, KVL_);
    mla_gemm_bf16<<<dim3((H_ * (NOPE_ + VHD_)) / BN, BS_ / BM), 256, 0, stream>>>(
        ckv_b, wkvb, kvb_raw, BS_, H_ * (NOPE_ + VHD_), KVL_);

    // ---- rope + head layout ----
    mla_rope_q<<<8192, 256, 0, stream>>>(q_raw, cosb, sinb, qb);
    mla_build_kv<<<8192, 256, 0, stream>>>(kvb_raw, kva_raw, cosb, sinb, kb, vb);

    // ---- attention ----
    mla_flash_attn<<<dim3(S_ / (16 * WPB), B_ * H_), 32 * WPB, 0, stream>>>(
        qb, kb, vb, attn_b, S_);

    // ---- output projection ----
    mla_gemm_bf16<<<dim3(DM_ / BN, BS_ / BM), 256, 0, stream>>>(
        attn_b, wo, out, BS_, DM_, H_ * VHD_);
}
